// ParallelAttention_45586782880438
// MI455X (gfx1250) — matrix operand order repaired and checked by reading, not compiled, not run on hardware
//
#include <hip/hip_runtime.h>
#include <math.h>

// Problem constants (match reference)
#define S_LEN   2048
#define D_MODEL 1024
#define NH      16
#define DKV     64
#define NBKT    32

typedef __attribute__((ext_vector_type(16))) __bf16    v16bf;
typedef __attribute__((ext_vector_type(8)))  float     v8f;
typedef __attribute__((ext_vector_type(4)))  unsigned  uint32x4;
typedef __attribute__((ext_vector_type(8)))  unsigned  uint32x8;

__device__ __forceinline__ v8f wmma_bf16(v16bf a, v16bf b, v8f c) {
    // D = A(16x32 bf16) * B(32x16 bf16) + C(16x16 f32)
    return __builtin_amdgcn_wmma_f32_16x16x32_bf16(
        /*neg_a=*/false, a, /*neg_b=*/false, b,
        /*c_mod=*/(short)0, c, /*reuse_a=*/false, /*reuse_b=*/false);
}

// CDNA5 async global->LDS copy (ASYNCcnt-tracked), 16 bytes per lane.
__device__ __forceinline__ void async_load_b128(unsigned lds_addr, const void* gaddr) {
    asm volatile("global_load_async_to_lds_b128 %0, %1, off"
                 :: "v"(lds_addr), "v"(gaddr)
                 : "memory");
}
__device__ __forceinline__ void wait_asynccnt0() {
    asm volatile("s_wait_asynccnt 0x0" ::: "memory");
}

// CDNA5 Tensor Data Mover: one 2-D tile DMA (global -> LDS), TENSORcnt-tracked.
// D# groups per CDNA5 ISA (2-group form, <=2D tensor).
__device__ __forceinline__ void tdm_load_2d(unsigned lds_addr, const void* gaddr,
                                            unsigned tensor_d0, unsigned tensor_d1,
                                            unsigned d0_stride,
                                            unsigned tile_d0, unsigned tile_d1,
                                            unsigned pad_interval_code,
                                            unsigned pad_amount_code,
                                            unsigned pad_enable) {
    unsigned long long ga = (unsigned long long)(size_t)gaddr;
    uint32x4 g0;
    g0[0] = 1u;                                             // count=1, user mode
    g0[1] = lds_addr;                                       // lds_addr [63:32]
    g0[2] = (unsigned)ga;                                   // global_addr [95:64]
    g0[3] = (unsigned)((ga >> 32) & 0x1FFFFFFu) | (2u << 30); // addr[56:32] | type=2
    uint32x8 g1;
    g1[0] = (1u << 16)                                      // data_size = 2 bytes
          | (pad_enable << 20)
          | (pad_interval_code << 22)
          | (pad_amount_code << 25);
    g1[1] = (tensor_d0 & 0xFFFFu) << 16;                    // tensor_dim0 [63:48]
    g1[2] = (tensor_d0 >> 16) | ((tensor_d1 & 0xFFFFu) << 16);
    g1[3] = (tensor_d1 >> 16) | ((tile_d0 & 0xFFFFu) << 16); // tile_dim0 [127:112]
    g1[4] = (tile_d1 & 0xFFFFu);                            // tile_dim1, tile_dim2=0
    g1[5] = d0_stride;                                      // tensor_dim0_stride[31:0]
    g1[6] = 0u;                                             // stride hi / dim1_stride
    g1[7] = 0u;
    asm volatile("tensor_load_to_lds %0, %1" :: "s"(g0), "s"(g1) : "memory");
}

// ---------------------------------------------------------------------------
// Kernel 1: T5 relative-position bias table  bias[h][ (k-q) + S-1 ]
// ---------------------------------------------------------------------------
__global__ __launch_bounds__(256)
void bias_table_kernel(const float* __restrict__ emb, float* __restrict__ bias_tab) {
    const int total = NH * (2 * S_LEN - 1);
    int idx = blockIdx.x * 256 + threadIdx.x;
    if (idx >= total) return;
    int h  = idx / (2 * S_LEN - 1);
    int rp = idx % (2 * S_LEN - 1) - (S_LEN - 1);   // k - q
    int bucket = (rp > 0) ? (NBKT / 2) : 0;          // nb = 16
    int arp = (rp < 0) ? -rp : rp;
    const int max_exact = 8;                         // nb/2
    if (arp < max_exact) {
        bucket += arp;
    } else {
        float lg = __logf((float)arp / (float)max_exact) *
                   (8.0f / __logf(128.0f / (float)max_exact));
        int lb = max_exact + (int)lg;
        bucket += (lb < 15) ? lb : 15;
    }
    bias_tab[idx] = emb[bucket * NH + h];
}

// ---------------------------------------------------------------------------
// Kernel 2: QKV projection  [4096,1024] x [1024,3072] -> bf16 [B,H,S,64] heads
// Block 128x128, 8 waves (4Mx2N), per-wave 32x64 => 8 WMMAs per K-step.
// ---------------------------------------------------------------------------
__global__ __launch_bounds__(256)
void qkv_gemm_kernel(const float* __restrict__ A, const float* __restrict__ W,
                     __bf16* __restrict__ qws, __bf16* __restrict__ kws,
                     __bf16* __restrict__ vws) {
    __shared__ __attribute__((aligned(16))) __bf16 As[128][40];
    __shared__ __attribute__((aligned(16))) __bf16 BsT[128][40];
    const int t    = threadIdx.x;
    const int lane = t & 31;
    const int wave = t >> 5;
    const int wm   = wave & 3;
    const int wn   = wave >> 2;
    const int hi   = lane >> 4;
    const int ln   = lane & 15;
    const int tileM = blockIdx.y * 128;
    const int tileN = blockIdx.x * 128;
    const int ldw  = 3 * D_MODEL;
    v8f acc[2][4] = {};

    for (int kt = 0; kt < D_MODEL; kt += 32) {
        __syncthreads();
        #pragma unroll
        for (int i = 0; i < 4; ++i) {
            int c   = t + i * 256;
            int row = c >> 3;
            int col = (c & 7) * 4;
            float4 v = *reinterpret_cast<const float4*>(
                A + (size_t)(tileM + row) * D_MODEL + kt + col);
            As[row][col + 0] = (__bf16)v.x; As[row][col + 1] = (__bf16)v.y;
            As[row][col + 2] = (__bf16)v.z; As[row][col + 3] = (__bf16)v.w;
        }
        #pragma unroll
        for (int i = 0; i < 4; ++i) {
            int task = t + i * 256;
            int n  = task & 127;
            int kc = task >> 7;
            union { uint2 u; __bf16 h[4]; } pk;
            #pragma unroll
            for (int j = 0; j < 4; ++j)
                pk.h[j] = (__bf16)W[(size_t)(kt + kc * 4 + j) * ldw + tileN + n];
            *reinterpret_cast<uint2*>(&BsT[n][kc * 4]) = pk.u;
        }
        __syncthreads();

        v16bf afr[2], bfr[4];
        #pragma unroll
        for (int i = 0; i < 2; ++i) {
            int m = wm * 32 + i * 16 + ln;
            #pragma unroll
            for (int j = 0; j < 8; ++j) {
                afr[i][j]     = As[m][hi * 8 + j];
                afr[i][j + 8] = As[m][hi * 8 + 16 + j];
            }
        }
        #pragma unroll
        for (int j = 0; j < 4; ++j) {
            int n = wn * 64 + j * 16 + ln;
            #pragma unroll
            for (int jj = 0; jj < 16; ++jj)
                bfr[j][jj] = BsT[n][hi * 8 + jj + (jj & 8)];
        }
        #pragma unroll
        for (int i = 0; i < 2; ++i)
            #pragma unroll
            for (int j = 0; j < 4; ++j)
                acc[i][j] = wmma_bf16(afr[i], bfr[j], acc[i][j]);
    }

    // Scatter to [B, H, S, 64] bf16 head layout.
    // tileM is 128-aligned and 2048 % 128 == 0 => batch index is block-constant.
    const int which  = tileN / D_MODEL;
    const int bconst = tileM >> 11;
    const int sbase  = tileM & (S_LEN - 1);
    __bf16* dst = (which == 0) ? qws : (which == 1) ? kws : vws;
    #pragma unroll
    for (int i = 0; i < 2; ++i) {
        #pragma unroll
        for (int j = 0; j < 4; ++j) {
            int colbase = (tileN % D_MODEL) + wn * 64 + j * 16;
            int h = colbase / DKV;
            int d = colbase % DKV + ln;
            size_t base = (((size_t)(bconst * NH + h)) * S_LEN
                           + sbase + wm * 32 + i * 16 + 8 * hi) * DKV + d;
            #pragma unroll
            for (int r = 0; r < 8; ++r)
                dst[base + (size_t)r * DKV] = (__bf16)acc[i][j][r];
        }
    }
}

// ---------------------------------------------------------------------------
// Kernel 3: flash attention per (b,h).  Block: 128 q-rows, 8 waves x 16 rows.
// K tile staged via Tensor Data Mover (2-D tile DMA with LDS row padding);
// V tile staged transposed with plain loads.
// ---------------------------------------------------------------------------
__global__ __launch_bounds__(256)
void attn_kernel(const __bf16* __restrict__ qws, const __bf16* __restrict__ kws,
                 const __bf16* __restrict__ vws, const float* __restrict__ mask,
                 const float* __restrict__ bias_tab, __bf16* __restrict__ ctx) {
    __shared__ __attribute__((aligned(16))) __bf16 Ks[32][72];   // [k][d], 144B rows
    __shared__ __attribute__((aligned(16))) __bf16 Vt[64][40];   // [d][k] transposed
    __shared__ __attribute__((aligned(16))) __bf16 pbuf[8][16][40];

    const int bh   = blockIdx.x;
    const int b    = bh / NH;
    const int h    = bh % NH;
    const int t    = threadIdx.x;
    const int wave = t >> 5;
    const int lane = t & 31;
    const int hi   = lane >> 4;
    const int ln   = lane & 15;
    const int q0   = blockIdx.y * 128 + wave * 16;

    const __bf16* Q  = qws + (size_t)bh * S_LEN * DKV;
    const __bf16* K  = kws + (size_t)bh * S_LEN * DKV;
    const __bf16* V  = vws + (size_t)bh * S_LEN * DKV;
    const float*  MK = mask + (size_t)b * S_LEN * S_LEN;
    const float*  BT = bias_tab + (size_t)h * (2 * S_LEN - 1);

    const int krow = t >> 3;           // V staging: row of the 32-row tile
    const int kchk = t & 7;            // 16-byte chunk within 128B row
    const unsigned ldsKsBase = (unsigned)(size_t)&Ks[0][0];

    // Q A-fragments (whole K loop), d split 0..31 / 32..63
    v16bf aq[2];
    {
        int m = q0 + ln;
        #pragma unroll
        for (int f = 0; f < 2; ++f) {
            #pragma unroll
            for (int j = 0; j < 8; ++j) {
                aq[f][j]     = Q[(size_t)m * DKV + f * 32 + hi * 8 + j];
                aq[f][j + 8] = Q[(size_t)m * DKV + f * 32 + hi * 8 + 16 + j];
            }
        }
    }

    v8f o[4] = {};
    float mrun[8], srun[8];
    #pragma unroll
    for (int r = 0; r < 8; ++r) { mrun[r] = -1e30f; srun[r] = 0.0f; }

    for (int kt = 0; kt < S_LEN; kt += 32) {
        __syncthreads();   // prior iteration's readers done
        // K tile: one TDM descriptor per block.  32x64 bf16 tile, 128B rows
        // padded to 144B in LDS (pad after 32 DWORDs by 4 DWORDs).
        if (wave == 0) {
            tdm_load_2d(ldsKsBase, K + (size_t)kt * DKV,
                        /*tensor_d0=*/DKV, /*tensor_d1=*/(unsigned)(S_LEN - kt),
                        /*d0_stride=*/DKV,
                        /*tile_d0=*/DKV, /*tile_d1=*/32,
                        /*pad_interval=*/4u /*32 DW*/, /*pad_amount=*/3u /*4 DW*/,
                        /*pad_enable=*/1u);
        }
        // V tile: coalesced read, transposed scatter into Vt[d][k]
        union { uint4 u; __bf16 hx[8]; } vv;
        vv.u = *reinterpret_cast<const uint4*>(V + (size_t)(kt + krow) * DKV + kchk * 8);
        #pragma unroll
        for (int j = 0; j < 8; ++j) Vt[kchk * 8 + j][krow] = vv.hx[j];
        if (wave == 0) __builtin_amdgcn_s_wait_tensorcnt(0);
        __syncthreads();   // staging visible to all waves

        // ---- scores S = Q * K^T (16 x 32) : 4 WMMAs ----
        v16bf bk[2][2];             // [d-half][k n-tile]
        #pragma unroll
        for (int f = 0; f < 2; ++f) {
            #pragma unroll
            for (int jn = 0; jn < 2; ++jn) {
                #pragma unroll
                for (int j = 0; j < 16; ++j)
                    bk[f][jn][j] = Ks[jn * 16 + ln][f * 32 + hi * 8 + j + (j & 8)];
            }
        }
        v8f sc[2] = {};
        #pragma unroll
        for (int jn = 0; jn < 2; ++jn) {
            sc[jn] = wmma_bf16(aq[0], bk[0][jn], sc[jn]);
            sc[jn] = wmma_bf16(aq[1], bk[1][jn], sc[jn]);
        }

        // ---- mask + bias + online softmax ----
        float p0[8], p1[8];
        #pragma unroll
        for (int r = 0; r < 8; ++r) {
            int qrow = q0 + r + 8 * hi;
            int kc0  = kt + ln;
            int kc1  = kt + 16 + ln;
            float mk0 = MK[(size_t)qrow * S_LEN + kc0];
            float mk1 = MK[(size_t)qrow * S_LEN + kc1];
            float bs0 = BT[kc0 - qrow + (S_LEN - 1)];
            float bs1 = BT[kc1 - qrow + (S_LEN - 1)];
            float v0 = sc[0][r] * 0.125f;
            float v1 = sc[1][r] * 0.125f;
            v0 = v0 * mk0 + (-10000.0f * (1.0f - mk0) + bs0);
            v1 = v1 * mk1 + (-10000.0f * (1.0f - mk1) + bs1);

            float mx = fmaxf(v0, v1);
            mx = fmaxf(mx, __shfl_xor(mx, 1, 32));
            mx = fmaxf(mx, __shfl_xor(mx, 2, 32));
            mx = fmaxf(mx, __shfl_xor(mx, 4, 32));
            mx = fmaxf(mx, __shfl_xor(mx, 8, 32));
            float mnew = fmaxf(mrun[r], mx);

            float e0 = __expf(v0 - mnew);
            float e1 = __expf(v1 - mnew);
            p0[r] = e0; p1[r] = e1;
            float rs = e0 + e1;
            rs += __shfl_xor(rs, 1, 32);
            rs += __shfl_xor(rs, 2, 32);
            rs += __shfl_xor(rs, 4, 32);
            rs += __shfl_xor(rs, 8, 32);

            float alpha = __expf(mrun[r] - mnew);
            srun[r] = srun[r] * alpha + rs;
            mrun[r] = mnew;
            #pragma unroll
            for (int dt = 0; dt < 4; ++dt) o[dt][r] *= alpha;
        }

        // ---- transpose P (C-layout -> A-layout) through per-wave LDS ----
        #pragma unroll
        for (int r = 0; r < 8; ++r) {
            int m = r + 8 * hi;
            pbuf[wave][m][ln]      = (__bf16)p0[r];
            pbuf[wave][m][16 + ln] = (__bf16)p1[r];
        }
        v16bf ap;
        #pragma unroll
        for (int j = 0; j < 8; ++j) {
            ap[j]     = pbuf[wave][ln][hi * 8 + j];
            ap[j + 8] = pbuf[wave][ln][hi * 8 + 16 + j];
        }

        // ---- O += P * V : 4 WMMAs, B-fragments from Vt rows in AMD table k order ----
        #pragma unroll
        for (int dt = 0; dt < 4; ++dt) {
            v16bf bv;
            #pragma unroll
            for (int j = 0; j < 16; ++j)
                bv[j] = Vt[dt * 16 + ln][hi * 8 + j + (j & 8)];
            o[dt] = wmma_bf16(ap, bv, o[dt]);
        }
    }

    // finalize: divide by denom, store ctx bf16 [B, S, H*64]
    #pragma unroll
    for (int dt = 0; dt < 4; ++dt) {
        size_t base = ((size_t)b * S_LEN + q0 + 8 * hi) * D_MODEL
                      + h * DKV + dt * 16 + ln;
        #pragma unroll
        for (int r = 0; r < 8; ++r)
            ctx[base + (size_t)r * D_MODEL] = (__bf16)(o[dt][r] / srun[r]);
    }
}

// ---------------------------------------------------------------------------
// Kernel 4: output projection  ctx(bf16)[4096,1024] x W_o[1024,1024] -> f32
// A tile staged via CDNA5 async global->LDS (pure bf16 identity copy).
// ---------------------------------------------------------------------------
__global__ __launch_bounds__(256)
void out_gemm_kernel(const __bf16* __restrict__ Actx, const float* __restrict__ W,
                     float* __restrict__ out) {
    __shared__ __attribute__((aligned(16))) __bf16 As[128][40];
    __shared__ __attribute__((aligned(16))) __bf16 BsT[128][40];
    const int t    = threadIdx.x;
    const int lane = t & 31;
    const int wave = t >> 5;
    const int wm   = wave & 3;
    const int wn   = wave >> 2;
    const int hi   = lane >> 4;
    const int ln   = lane & 15;
    const int tileM = blockIdx.y * 128;
    const int tileN = blockIdx.x * 128;
    v8f acc[2][4] = {};

    for (int kt = 0; kt < D_MODEL; kt += 32) {
        __syncthreads();
        // A tile: bf16 identity copy via async global->LDS (2 x b128 / thread)
        #pragma unroll
        for (int i = 0; i < 2; ++i) {
            int c   = t + i * 256;
            int row = c >> 2;
            int off = (c & 3) * 8;
            async_load_b128((unsigned)(size_t)&As[row][off],
                            Actx + (size_t)(tileM + row) * D_MODEL + kt + off);
        }
        // B tile: transposed reads of W_o, packed b64 stores
        #pragma unroll
        for (int i = 0; i < 4; ++i) {
            int task = t + i * 256;
            int n  = task & 127;
            int kc = task >> 7;
            union { uint2 u; __bf16 h[4]; } pk;
            #pragma unroll
            for (int j = 0; j < 4; ++j)
                pk.h[j] = (__bf16)W[(size_t)(kt + kc * 4 + j) * D_MODEL + tileN + n];
            *reinterpret_cast<uint2*>(&BsT[n][kc * 4]) = pk.u;
        }
        wait_asynccnt0();
        __syncthreads();

        v16bf afr[2], bfr[4];
        #pragma unroll
        for (int i = 0; i < 2; ++i) {
            int m = wm * 32 + i * 16 + ln;
            #pragma unroll
            for (int j = 0; j < 8; ++j) {
                afr[i][j]     = As[m][hi * 8 + j];
                afr[i][j + 8] = As[m][hi * 8 + 16 + j];
            }
        }
        #pragma unroll
        for (int j = 0; j < 4; ++j) {
            int n = wn * 64 + j * 16 + ln;
            #pragma unroll
            for (int jj = 0; jj < 16; ++jj)
                bfr[j][jj] = BsT[n][hi * 8 + jj + (jj & 8)];
        }
        #pragma unroll
        for (int i = 0; i < 2; ++i)
            #pragma unroll
            for (int j = 0; j < 4; ++j)
                acc[i][j] = wmma_bf16(afr[i], bfr[j], acc[i][j]);
    }

    #pragma unroll
    for (int i = 0; i < 2; ++i) {
        #pragma unroll
        for (int j = 0; j < 4; ++j) {
            size_t base = (size_t)(tileM + wm * 32 + i * 16 + 8 * hi) * D_MODEL
                          + tileN + wn * 64 + j * 16 + ln;
            #pragma unroll
            for (int r = 0; r < 8; ++r)
                out[base + (size_t)r * D_MODEL] = acc[i][j][r];
        }
    }
}

// ---------------------------------------------------------------------------
// Launch
// ---------------------------------------------------------------------------
extern "C" void kernel_launch(void* const* d_in, const int* in_sizes, int n_in,
                              void* d_out, int out_size, void* d_ws, size_t ws_size,
                              hipStream_t stream) {
    const float* hidden = (const float*)d_in[0];
    const float* mask   = (const float*)d_in[1];
    const float* W_qkv  = (const float*)d_in[2];
    const float* W_o    = (const float*)d_in[3];
    const float* emb    = (const float*)d_in[4];
    float* out = (float*)d_out;

    char* p = (char*)d_ws;
    auto carve = [&](size_t bytes) {
        void* q = (void*)p;
        p += (bytes + 255) & ~(size_t)255;
        return q;
    };
    const size_t headBytes = (size_t)2 * NH * S_LEN * DKV * sizeof(__bf16); // 8 MB
    __bf16* qws = (__bf16*)carve(headBytes);
    __bf16* kws = (__bf16*)carve(headBytes);
    __bf16* vws = (__bf16*)carve(headBytes);
    __bf16* ctx = (__bf16*)carve((size_t)2 * S_LEN * D_MODEL * sizeof(__bf16));
    float* bias_tab = (float*)carve((size_t)NH * (2 * S_LEN - 1) * sizeof(float));

    bias_table_kernel<<<dim3((NH * (2 * S_LEN - 1) + 255) / 256), dim3(256), 0, stream>>>(
        emb, bias_tab);
    qkv_gemm_kernel<<<dim3(3 * D_MODEL / 128, 2 * S_LEN / 128), dim3(256), 0, stream>>>(
        hidden, W_qkv, qws, kws, vws);
    attn_kernel<<<dim3(2 * NH, S_LEN / 128), dim3(256), 0, stream>>>(
        qws, kws, vws, mask, bias_tab, ctx);
    out_gemm_kernel<<<dim3(D_MODEL / 128, 2 * S_LEN / 128), dim3(256), 0, stream>>>(
        ctx, W_o, out);
}
